// BonsaiStumpSAGENet_86088324481903
// MI455X (gfx1250) — compile-verified, matrix operand order checked by reading, not run on hardware
//
#include <hip/hip_runtime.h>
#include <hip/hip_bf16.h>
#include <float.h>

// ---------------------------------------------------------------------------
// BonsaiStumpSAGENet on gfx1250 (MI455X).
//
// d_in layout assumption:
//   [0] x_bonsai [N,4] f32       [1] edge_index_bonsai [2,E] i32
//   [2] batch_bonsai [N] i32     [3] x_stump [N,4] f32
//   [4] edge_index_stump [2,E]   [5] batch_stump [N] i32
//   [6] ptr [G] i32
//   [7..] params flattened as a JAX pytree (dict keys ASCII-sorted, lists in
//   order).  Encoder leaf order (relative to tower base):
//     +0..+11 : convs[l].{Wl,Wr,bl} for l=0..3
//     +12..+19: lns[l].{b,g}        for l=0..3
//     +20 mlp_W1  +21 mlp_W2  +22 mlp_b  +23 mlp_b1  +24 mlp_b2  +25 mlp_g
//     +26 res_W   +27 res_b
//   Tower bases: bonsai=7, stump=41.  Readout: 35 ro_W1, 36 ro_W2, 37 ro_b,
//   38 ro_b1, 39 ro_b2, 40 ro_g.
// ---------------------------------------------------------------------------

typedef __attribute__((ext_vector_type(16))) _Float16 v16h;
typedef __attribute__((ext_vector_type(8)))  float    v8f;

#define NHID 64

// ---- WMMA fragment helpers (CDNA5 16-bit A/B layout, f32 C/D layout) ------
__device__ __forceinline__ v16h frag_ld(const _Float16* base, int stride, int k0, int lane) {
  int r  = lane & 15;                       // M (A) or N (B) row
  int kb = k0 + ((lane & 16) ? 8 : 0);      // K base for this half-wave
  const _Float16* p = base + r * stride + kb;
  v16h f;
#pragma unroll
  for (int i = 0; i < 8; ++i) f[i] = p[i];
#pragma unroll
  for (int i = 0; i < 8; ++i) f[8 + i] = p[16 + i];
  return f;
}

__device__ __forceinline__ float silu(float x) { return x / (1.0f + __expf(-x)); }

__device__ __forceinline__ void atomicMaxF(float* addr, float v) {
  if (v >= 0.0f) atomicMax((int*)addr, __float_as_int(v));
  else           atomicMin((unsigned int*)addr, __float_as_uint(v));
}

// ---- tiny utility kernels -------------------------------------------------
__global__ void k_fillf(float* p, float v, int n) {
  int i = blockIdx.x * blockDim.x + threadIdx.x;
  if (i < n) p[i] = v;
}

__global__ void k_cvt_f16(const float* __restrict__ s, _Float16* __restrict__ d, int n) {
  int i = blockIdx.x * blockDim.x + threadIdx.x;
  if (i < n) d[i] = (_Float16)s[i];
}

// Pack [Wl | Wr] (row-major per out) into f16 [64][Kpad], zero padded.
__global__ void k_pack_wcat(const float* __restrict__ Wl, const float* __restrict__ Wr,
                            _Float16* __restrict__ out, int cin, int Kpad) {
  int i = blockIdx.x * blockDim.x + threadIdx.x;
  if (i >= NHID * Kpad) return;
  int o = i / Kpad, k = i % Kpad;
  float v = 0.0f;
  if (k < 2 * cin)       v = Wl[o * 2 * cin + k];
  else if (k < 3 * cin)  v = Wr[o * cin + (k - 2 * cin)];
  out[i] = (_Float16)v;
}

// ---- edge phase (L2-resident scatter/gather) ------------------------------
__global__ void k_degree(const int* __restrict__ dst, float* __restrict__ cnt, int E) {
  int e = blockIdx.x * blockDim.x + threadIdx.x;
  if (e < E) atomicAdd(&cnt[dst[e]], 1.0f);
}

__global__ void k_edge4(const int* __restrict__ src, const int* __restrict__ dst,
                        const float* __restrict__ x, float* __restrict__ sum,
                        float* __restrict__ mx, int E) {
  int e = blockIdx.x * blockDim.x + threadIdx.x;
  if (e >= E) return;
  int s = src[e], d = dst[e];
  float4 v = *(const float4*)(x + (size_t)s * 4);
  float* ps = sum + (size_t)d * 4;
  float* pm = mx  + (size_t)d * 4;
  atomicAdd(ps + 0, v.x); atomicAdd(ps + 1, v.y);
  atomicAdd(ps + 2, v.z); atomicAdd(ps + 3, v.w);
  atomicMaxF(pm + 0, v.x); atomicMaxF(pm + 1, v.y);
  atomicMaxF(pm + 2, v.z); atomicMaxF(pm + 3, v.w);
}

__global__ void k_edge64(const int* __restrict__ src, const int* __restrict__ dst,
                         const float* __restrict__ h, float* __restrict__ sum,
                         float* __restrict__ mx, int E) {
  int t = blockIdx.x * blockDim.x + threadIdx.x;   // E*16 threads, 16 per edge
  if (t >= E * 16) return;
  int e = t >> 4, c = (t & 15) << 2;
  int s = src[e], d = dst[e];
  float4 v = *(const float4*)(h + (size_t)s * NHID + c);
  float* ps = sum + (size_t)d * NHID + c;
  float* pm = mx  + (size_t)d * NHID + c;
  atomicAdd(ps + 0, v.x); atomicAdd(ps + 1, v.y);
  atomicAdd(ps + 2, v.z); atomicAdd(ps + 3, v.w);
  atomicMaxF(pm + 0, v.x); atomicMaxF(pm + 1, v.y);
  atomicMaxF(pm + 2, v.z); atomicMaxF(pm + 3, v.w);
}

// ---- fused SAGE node update: u=[max|mean|root] -> WMMA -> LN -> +res -> SiLU
// block = 128 threads (4 waves) = 16 nodes.  Wave w computes out cols 16w..16w+15.
__global__ void k_node_update(const float* __restrict__ sum, const float* __restrict__ mx,
                              const float* __restrict__ cnt, const float* __restrict__ hin,
                              const _Float16* __restrict__ Wh,   // [64][Kpad] f16
                              const float* __restrict__ bl,
                              const float* __restrict__ lng, const float* __restrict__ lnb,
                              const float* __restrict__ resW, const float* __restrict__ resb,
                              const float* __restrict__ xroot,
                              float* __restrict__ hout,
                              int cin, int Kpad, int layer0) {
  __shared__ _Float16 sU[16 * 192];   // node tile, f16, stride Kpad
  __shared__ float    sH[16 * NHID];  // pre-LN outputs
  int n0   = blockIdx.x * 16;
  int wave = threadIdx.x >> 5;
  int lane = threadIdx.x & 31;

  for (int idx = threadIdx.x; idx < 16 * Kpad; idx += 128) {
    int n = idx / Kpad, k = idx % Kpad;
    int g = n0 + n;
    float c = cnt[g];
    float v = 0.0f;
    if (k < cin)            v = (c > 0.0f) ? mx[(size_t)g * cin + k] : 0.0f;
    else if (k < 2 * cin)   v = sum[(size_t)g * cin + (k - cin)] / fmaxf(c, 1.0f);
    else if (k < 3 * cin)   v = hin[(size_t)g * cin + (k - 2 * cin)];
    sU[idx] = (_Float16)v;
  }
  __syncthreads();

  v8f acc = {};
  for (int k0 = 0; k0 < Kpad; k0 += 32) {
    v16h a = frag_ld(sU, Kpad, k0, lane);
    v16h b = frag_ld(Wh + wave * 16 * Kpad, Kpad, k0, lane);
    acc = __builtin_amdgcn_wmma_f32_16x16x32_f16(false, a, false, b, (short)0, acc,
                                                 false, false);
  }
  {
    int moff = (lane & 16) ? 8 : 0, nc = lane & 15;
#pragma unroll
    for (int v = 0; v < 8; ++v) sH[(v + moff) * NHID + wave * 16 + nc] = acc[v];
  }
  __syncthreads();

  if (threadIdx.x < 16) {
    int n = threadIdx.x, g = n0 + n;
    float s = 0.0f, s2 = 0.0f;
    for (int o = 0; o < NHID; ++o) {
      float v = sH[n * NHID + o] + bl[o];
      s += v; s2 += v * v;
    }
    float m  = s / NHID;
    float rs = rsqrtf(s2 / NHID - m * m + 1e-5f);
    for (int o = 0; o < NHID; ++o) {
      float v  = sH[n * NHID + o] + bl[o];
      float ln = (v - m) * rs * lng[o] + lnb[o];
      float res;
      if (layer0) {
        res = resb[o];
        for (int i = 0; i < 4; ++i) res += xroot[(size_t)g * 4 + i] * resW[o * 4 + i];
      } else {
        res = hin[(size_t)g * NHID + o];
      }
      hout[(size_t)g * NHID + o] = silu(ln + res);
    }
  }
}

// ---- fused MLP: silu(h@W1+b1) -> LN -> @W2+b2  (16 nodes / block, 4 waves) --
__global__ void k_mlp(const float* __restrict__ hin,
                      const _Float16* __restrict__ W1h,  // [256][64]
                      const float* __restrict__ b1, const float* __restrict__ g1,
                      const float* __restrict__ bln,
                      const _Float16* __restrict__ W2h,  // [64][256]
                      const float* __restrict__ b2,
                      float* __restrict__ hout) {
  __shared__ _Float16 sA[16 * 64];
  __shared__ float    sT[16 * 256];
  __shared__ _Float16 sTh[16 * 256];
  int n0   = blockIdx.x * 16;
  int wave = threadIdx.x >> 5;
  int lane = threadIdx.x & 31;

  for (int idx = threadIdx.x; idx < 16 * 64; idx += 128)
    sA[idx] = (_Float16)hin[(size_t)n0 * 64 + idx];
  __syncthreads();

  for (int j = wave; j < 16; j += 4) {        // 16 col tiles of 16
    v8f acc = {};
    for (int k0 = 0; k0 < 64; k0 += 32) {
      v16h a = frag_ld(sA, 64, k0, lane);
      v16h b = frag_ld(W1h + j * 16 * 64, 64, k0, lane);
      acc = __builtin_amdgcn_wmma_f32_16x16x32_f16(false, a, false, b, (short)0, acc,
                                                   false, false);
    }
    int moff = (lane & 16) ? 8 : 0, nc = lane & 15;
#pragma unroll
    for (int v = 0; v < 8; ++v) sT[(v + moff) * 256 + j * 16 + nc] = acc[v];
  }
  __syncthreads();

  for (int idx = threadIdx.x; idx < 16 * 256; idx += 128) {
    int k = idx & 255;
    sT[idx] = silu(sT[idx] + b1[k]);
  }
  __syncthreads();

  if (threadIdx.x < 16) {
    int n = threadIdx.x;
    float s = 0.0f, s2 = 0.0f;
    for (int k = 0; k < 256; ++k) { float v = sT[n * 256 + k]; s += v; s2 += v * v; }
    float m  = s / 256.0f;
    float rs = rsqrtf(s2 / 256.0f - m * m + 1e-5f);
    for (int k = 0; k < 256; ++k)
      sTh[n * 256 + k] = (_Float16)((sT[n * 256 + k] - m) * rs * g1[k] + bln[k]);
  }
  __syncthreads();

  {                                           // 4 col tiles, one per wave
    v8f acc = {};
    for (int k0 = 0; k0 < 256; k0 += 32) {
      v16h a = frag_ld(sTh, 256, k0, lane);
      v16h b = frag_ld(W2h + wave * 16 * 256, 256, k0, lane);
      acc = __builtin_amdgcn_wmma_f32_16x16x32_f16(false, a, false, b, (short)0, acc,
                                                   false, false);
    }
    int moff = (lane & 16) ? 8 : 0, nc = lane & 15;
#pragma unroll
    for (int v = 0; v < 8; ++v) {
      int g = n0 + v + moff;
      hout[(size_t)g * 64 + wave * 16 + nc] = acc[v] + b2[wave * 16 + nc];
    }
  }
}

// ---- pooling: [mean | max | sum] per graph --------------------------------
__global__ void k_pool(const float* __restrict__ h, const int* __restrict__ ptr,
                       int G, int N, float* __restrict__ pooled) {
  __shared__ float ss[256], sm[256];
  int g = blockIdx.x;
  int start = ptr[g];
  int end   = (g + 1 < G) ? ptr[g + 1] : N;
  int f = threadIdx.x & 63, ch = threadIdx.x >> 6;
  float s = 0.0f, m = -FLT_MAX;
  for (int i = start + ch; i < end; i += 4) {
    float v = h[(size_t)i * 64 + f];
    s += v; m = fmaxf(m, v);
  }
  ss[threadIdx.x] = s; sm[threadIdx.x] = m;
  __syncthreads();
  if (threadIdx.x < 64) {
    for (int c = 1; c < 4; ++c) { s += ss[c * 64 + f]; m = fmaxf(m, sm[c * 64 + f]); }
    float c = (float)(end - start);
    pooled[g * 192 + f]       = s / fmaxf(c, 1.0f);
    pooled[g * 192 + 64 + f]  = (c > 0.0f) ? m : 0.0f;
    pooled[g * 192 + 128 + f] = s;
  }
}

// ---- readout --------------------------------------------------------------
__global__ void k_read1(const float* __restrict__ pb, const float* __restrict__ ps,
                        const float* __restrict__ x, const int* __restrict__ ptr,
                        const float* __restrict__ W1, const float* __restrict__ b1,
                        float* __restrict__ z1, int G) {
  int idx = blockIdx.x * blockDim.x + threadIdx.x;
  if (idx >= G * 256) return;
  int g = idx / 256, o = idx % 256;
  const float* w = W1 + (size_t)o * 388;
  float acc = b1[o];
  for (int k = 0; k < 192; ++k) acc += pb[g * 192 + k] * w[k];
  for (int k = 0; k < 192; ++k) acc += ps[g * 192 + k] * w[192 + k];
  int r = ptr[g];
  for (int k = 0; k < 4; ++k)   acc += x[(size_t)r * 4 + k] * w[384 + k];
  z1[idx] = silu(acc);
}

__global__ void k_ln256(const float* __restrict__ z1, const float* __restrict__ gg,
                        const float* __restrict__ bb, float* __restrict__ zo) {
  __shared__ float red[256];
  int g = blockIdx.x, t = threadIdx.x;
  float v = z1[g * 256 + t];
  red[t] = v; __syncthreads();
  for (int s = 128; s > 0; s >>= 1) { if (t < s) red[t] += red[t + s]; __syncthreads(); }
  float mean = red[0] / 256.0f;
  __syncthreads();
  float d = v - mean;
  red[t] = d * d; __syncthreads();
  for (int s = 128; s > 0; s >>= 1) { if (t < s) red[t] += red[t + s]; __syncthreads(); }
  float rs = rsqrtf(red[0] / 256.0f + 1e-5f);
  zo[g * 256 + t] = d * rs * gg[t] + bb[t];
}

__global__ void k_read2(const float* __restrict__ zn, const float* __restrict__ W2,
                        const float* __restrict__ b2, float* __restrict__ out, int G) {
  int idx = blockIdx.x * blockDim.x + threadIdx.x;
  if (idx >= G * 2) return;
  int g = idx / 2, j = idx % 2;
  float acc = b2[j];
  for (int k = 0; k < 256; ++k) acc += zn[g * 256 + k] * W2[j * 256 + k];
  out[idx] = acc;
}

// ---------------------------------------------------------------------------
extern "C" void kernel_launch(void* const* d_in, const int* in_sizes, int n_in,
                              void* d_out, int out_size, void* d_ws, size_t ws_size,
                              hipStream_t stream) {
  const int N = in_sizes[0] / 4;
  const int E = in_sizes[1] / 2;
  const int G = in_sizes[6];

  // enc-relative leaf indices
  enum { L_C0WL = 0, L_C0WR, L_C0BL, L_C1WL, L_C1WR, L_C1BL, L_C2WL, L_C2WR, L_C2BL,
         L_C3WL, L_C3WR, L_C3BL, L_LN0B, L_LN0G, L_LN1B, L_LN1G, L_LN2B, L_LN2G,
         L_LN3B, L_LN3G, L_MW1, L_MW2, L_MB, L_MB1, L_MB2, L_MG, L_RESW, L_RESB };
  const int PB[2] = { 7, 41 };
  const int I_ROW1 = 35, I_ROW2 = 36, I_ROB = 37, I_ROB1 = 38, I_ROB2 = 39, I_ROG = 40;

  // workspace layout
  size_t off = 0;
  auto alloc = [&](size_t bytes) { size_t o = off; off = (off + bytes + 255) & ~(size_t)255; return o; };
  const size_t hbytes = (size_t)N * NHID * sizeof(float);
  size_t oHA   = alloc(hbytes);
  size_t oHB   = alloc(hbytes);
  size_t oSUM  = alloc(hbytes);
  size_t oMX   = alloc(hbytes);
  size_t oCNT  = alloc((size_t)N * sizeof(float));
  size_t oW16  = alloc(2 * 71680 * sizeof(_Float16));
  size_t oPOOL = alloc(2 * (size_t)G * 192 * sizeof(float));
  size_t oZ1   = alloc((size_t)G * 256 * sizeof(float));
  size_t oZ1N  = alloc((size_t)G * 256 * sizeof(float));
  (void)ws_size; (void)n_in;

  char* ws = (char*)d_ws;
  float* hA  = (float*)(ws + oHA);
  float* hB  = (float*)(ws + oHB);
  float* sum = (float*)(ws + oSUM);
  float* mxb = (float*)(ws + oMX);
  float* cnt = (float*)(ws + oCNT);
  float* z1  = (float*)(ws + oZ1);
  float* z1n = (float*)(ws + oZ1N);
  const int* ptr = (const int*)d_in[6];

  auto P = [&](int i) { return (const float*)d_in[i]; };
  const int TPB = 256;
  auto cdiv = [](long a, long b) { return (unsigned)((a + b - 1) / b); };

  for (int t = 0; t < 2; ++t) {
    const float* x  = P(t ? 3 : 0);
    const int* src  = (const int*)d_in[t ? 4 : 1];
    const int* dst  = src + E;
    const int pb = PB[t];
    _Float16* wf = (_Float16*)(ws + oW16) + (size_t)t * 71680;
    _Float16* wc[4] = { wf, wf + 2048, wf + 2048 + 12288, wf + 2048 + 2 * 12288 };
    _Float16* w1h = wf + 38912;
    _Float16* w2h = wf + 55296;
    float* pooled_t = (float*)(ws + oPOOL) + (size_t)t * G * 192;

    // pack weights -> f16 (tiny, L2-resident afterwards)
    k_pack_wcat<<<cdiv(64 * 32, TPB), TPB, 0, stream>>>(P(pb + L_C0WL), P(pb + L_C0WR), wc[0], 4, 32);
    for (int l = 1; l < 4; ++l)
      k_pack_wcat<<<cdiv(64 * 192, TPB), TPB, 0, stream>>>(P(pb + L_C0WL + 3 * l),
                                                           P(pb + L_C0WR + 3 * l), wc[l], 64, 192);
    k_cvt_f16<<<cdiv(256 * 64, TPB), TPB, 0, stream>>>(P(pb + L_MW1), w1h, 256 * 64);
    k_cvt_f16<<<cdiv(64 * 256, TPB), TPB, 0, stream>>>(P(pb + L_MW2), w2h, 64 * 256);

    // degree (shared by all layers)
    k_fillf<<<cdiv(N, TPB), TPB, 0, stream>>>(cnt, 0.0f, N);
    k_degree<<<cdiv(E, TPB), TPB, 0, stream>>>(dst, cnt, E);

    // layer 0 (cin=4)
    k_fillf<<<cdiv(N * 4, TPB), TPB, 0, stream>>>(sum, 0.0f, N * 4);
    k_fillf<<<cdiv(N * 4, TPB), TPB, 0, stream>>>(mxb, -FLT_MAX, N * 4);
    k_edge4<<<cdiv(E, TPB), TPB, 0, stream>>>(src, dst, x, sum, mxb, E);
    k_node_update<<<N / 16, 128, 0, stream>>>(sum, mxb, cnt, x, wc[0], P(pb + L_C0BL),
                                              P(pb + L_LN0G), P(pb + L_LN0B),
                                              P(pb + L_RESW), P(pb + L_RESB), x,
                                              hA, 4, 32, 1);
    float* cur = hA; float* nxt = hB;

    // layers 1..3 (cin=64)
    for (int l = 1; l < 4; ++l) {
      k_fillf<<<cdiv((long)N * NHID, TPB), TPB, 0, stream>>>(sum, 0.0f, N * NHID);
      k_fillf<<<cdiv((long)N * NHID, TPB), TPB, 0, stream>>>(mxb, -FLT_MAX, N * NHID);
      k_edge64<<<cdiv((long)E * 16, TPB), TPB, 0, stream>>>(src, dst, cur, sum, mxb, E);
      k_node_update<<<N / 16, 128, 0, stream>>>(sum, mxb, cnt, cur, wc[l],
                                                P(pb + L_C0BL + 3 * l),
                                                P(pb + L_LN0G + 2 * l), P(pb + L_LN0B + 2 * l),
                                                nullptr, nullptr, nullptr,
                                                nxt, 64, 192, 0);
      float* tmp = cur; cur = nxt; nxt = tmp;
    }

    // MLP head -> henc (the free buffer)
    k_mlp<<<N / 16, 128, 0, stream>>>(cur, w1h, P(pb + L_MB1), P(pb + L_MG), P(pb + L_MB),
                                      w2h, P(pb + L_MB2), nxt);
    // pooling
    k_pool<<<G, 256, 0, stream>>>(nxt, ptr, G, N, pooled_t);
  }

  // readout
  const float* pooled_b = (const float*)(ws + oPOOL);
  const float* pooled_s = pooled_b + (size_t)G * 192;
  k_read1<<<cdiv((long)G * 256, TPB), TPB, 0, stream>>>(pooled_b, pooled_s, P(0), ptr,
                                                        P(I_ROW1), P(I_ROB1), z1, G);
  k_ln256<<<G, 256, 0, stream>>>(z1, P(I_ROG), P(I_ROB), z1n);
  k_read2<<<cdiv((long)G * 2, TPB), TPB, 0, stream>>>(z1n, P(I_ROW2), P(I_ROB2),
                                                      (float*)d_out, G);
  (void)out_size;
}
